// GaussianModel_88562225643529
// MI455X (gfx1250) — matrix operand order repaired and checked by reading, not compiled, not run on hardware
//
#include <hip/hip_runtime.h>

// ---------------------------------------------------------------------------
// MI455X (gfx1250) Gaussian splat renderer.
//   * compute-bound (0.7 GFLOP + 33M exp, ~1MB memory traffic) -> keep all
//     per-gaussian data resident in LDS/L2, use WMMA f16 for the blend matmul.
//   * wave32: one wave owns 16 pixels; 32 sorted gaussians per chunk ->
//     one v_wmma_f32_16x16x32_f16 per chunk; the serial transmittance is
//     handled by per-lane 8-long prefix products + one shfl_xor(16) exchange
//     (matches the gfx1250 A-matrix lane split of each row's K dimension).
//   * chunk staging uses gfx1250 GLOBAL_LOAD_ASYNC_TO_LDS_B128 double
//     buffering (ASYNCcnt + s_wait_asynccnt) when the builtins exist,
//     else vectorized b128 sync copies.
// ---------------------------------------------------------------------------

typedef _Float16 v16h __attribute__((ext_vector_type(16)));
typedef float    v8f  __attribute__((ext_vector_type(8)));

#if defined(__AMDGCN__) && \
    __has_builtin(__builtin_amdgcn_global_load_async_to_lds_b128) && \
    __has_builtin(__builtin_amdgcn_s_wait_asynccnt)
#define HAVE_ASYNC_LDS 1
// builtin parameter types (from clang diagnostic): generic-vector int4 with
// addrspace(1) source / addrspace(3) LDS destination
typedef int v4i_vs __attribute__((vector_size(16)));
typedef __attribute__((address_space(1))) v4i_vs* gptr128_t;
typedef __attribute__((address_space(3))) v4i_vs* lptr128_t;
#else
#define HAVE_ASYNC_LDS 0
#endif

#define SH_C0 0.28209479177387814f
#define SH_C1 0.4886025119029199f

// ---------------------------------------------------------------------------
// Kernel 1: per-gaussian preprocess (one thread per gaussian)
// ---------------------------------------------------------------------------
__global__ void gs_preprocess(const float* __restrict__ xyz,
                              const float* __restrict__ scaling,
                              const float* __restrict__ rotation,
                              const float* __restrict__ opacity,
                              const float* __restrict__ fdc,
                              const float* __restrict__ frest,
                              const float* __restrict__ view,
                              const float* __restrict__ proj,
                              const float* __restrict__ campos,
                              const int* __restrict__ Hp,
                              const int* __restrict__ Wp,
                              float* __restrict__ tz_key,
                              float4* __restrict__ geo,   // 2x float4 / gaussian
                              float4* __restrict__ col,   // 1x float4 / gaussian
                              float* __restrict__ radii_out,
                              int N)
{
  int g = blockIdx.x * blockDim.x + threadIdx.x;
  if (g >= N) return;

  float X = xyz[3*g+0], Y = xyz[3*g+1], Z = xyz[3*g+2];
  float sx = __expf(scaling[3*g+0]);
  float sy = __expf(scaling[3*g+1]);
  float sz = __expf(scaling[3*g+2]);
  float op = 1.0f / (1.0f + __expf(-opacity[g]));

  // quaternion -> rotation
  float qw = rotation[4*g+0], qx = rotation[4*g+1];
  float qy = rotation[4*g+2], qz = rotation[4*g+3];
  float qn = sqrtf(qw*qw + qx*qx + qy*qy + qz*qz);
  qn = fmaxf(qn, 1e-12f);
  qw /= qn; qx /= qn; qy /= qn; qz /= qn;
  float r00 = 1.f - 2.f*(qy*qy + qz*qz), r01 = 2.f*(qx*qy - qw*qz), r02 = 2.f*(qx*qz + qw*qy);
  float r10 = 2.f*(qx*qy + qw*qz), r11 = 1.f - 2.f*(qx*qx + qz*qz), r12 = 2.f*(qy*qz - qw*qx);
  float r20 = 2.f*(qx*qz - qw*qy), r21 = 2.f*(qy*qz + qw*qx), r22 = 1.f - 2.f*(qx*qx + qy*qy);

  // M = R * diag(s); Sigma = M M^T (symmetric)
  float m00=r00*sx, m01=r01*sy, m02=r02*sz;
  float m10=r10*sx, m11=r11*sy, m12=r12*sz;
  float m20=r20*sx, m21=r21*sy, m22=r22*sz;
  float s00 = m00*m00 + m01*m01 + m02*m02;
  float s01 = m00*m10 + m01*m11 + m02*m12;
  float s02 = m00*m20 + m01*m21 + m02*m22;
  float s11 = m10*m10 + m11*m11 + m12*m12;
  float s12 = m10*m20 + m11*m21 + m12*m22;
  float s22 = m20*m20 + m21*m21 + m22*m22;

  // view transform: t_j = xyz @ V[:3,:3] + V[3,:3]
  float t0 = X*view[0] + Y*view[4] + Z*view[8]  + view[12];
  float t1 = X*view[1] + Y*view[5] + Z*view[9]  + view[13];
  float tz = X*view[2] + Y*view[6] + Z*view[10] + view[14];

  float Wf = (float)(*Wp), Hf = (float)(*Hp);
  float fx = 0.5f * Wf, fy = 0.5f * Hf;              // TANFOV = 1
  float tzs = (fabsf(tz) > 1e-6f) ? tz : 1e-6f;
  const float lim = 1.3f;
  float txz = fminf(fmaxf(t0 / tzs, -lim), lim) * tzs;
  float tyz = fminf(fmaxf(t1 / tzs, -lim), lim) * tzs;
  float j00 = fx / tzs, j02 = -fx * txz / (tzs * tzs);
  float j11 = fy / tzs, j12 = -fy * tyz / (tzs * tzs);

  // Tm = J @ V[:3,:3]^T   (Wr[k][c] = view[c*4+k])
  float T00 = j00*view[0]  + j02*view[2];
  float T01 = j00*view[4]  + j02*view[6];
  float T02 = j00*view[8]  + j02*view[10];
  float T10 = j11*view[1]  + j12*view[2];
  float T11 = j11*view[5]  + j12*view[6];
  float T12 = j11*view[9]  + j12*view[10];

  // cov2 = Tm Sigma Tm^T
  float u0 = s00*T00 + s01*T01 + s02*T02;
  float u1 = s01*T00 + s11*T01 + s12*T02;
  float u2 = s02*T00 + s12*T01 + s22*T02;
  float w0 = s00*T10 + s01*T11 + s02*T12;
  float w1 = s01*T10 + s11*T11 + s12*T12;
  float w2 = s02*T10 + s12*T11 + s22*T12;
  float a = T00*u0 + T01*u1 + T02*u2 + 0.3f;
  float b = T10*u0 + T11*u1 + T12*u2;
  float c = T10*w0 + T11*w1 + T12*w2 + 0.3f;

  float det = a*c - b*b;
  bool valid = (tz > 0.2f) && (det > 0.0f);
  float inv_det = 1.0f / ((det > 0.0f) ? det : 1.0f);
  float conA =  c * inv_det;
  float conB = -b * inv_det;
  float conC =  a * inv_det;
  float mid = 0.5f * (a + c);
  float lam = mid + sqrtf(fmaxf(mid*mid - det, 0.1f));
  radii_out[g] = valid ? ceilf(3.0f * sqrtf(lam)) : 0.0f;

  // projection to screen mean
  float p0 = X*proj[0] + Y*proj[4] + Z*proj[8]  + proj[12];
  float p1 = X*proj[1] + Y*proj[5] + Z*proj[9]  + proj[13];
  float p3 = X*proj[3] + Y*proj[7] + Z*proj[11] + proj[15];
  float pw = 1.0f / (p3 + 1e-7f);
  float mx = ((p0*pw + 1.0f) * Wf - 1.0f) * 0.5f;
  float my = ((p1*pw + 1.0f) * Hf - 1.0f) * 0.5f;

  // SH degree-3 color
  float dx = X - campos[0], dy = Y - campos[1], dz = Z - campos[2];
  float dn = fmaxf(sqrtf(dx*dx + dy*dy + dz*dz), 1e-12f);
  float x = dx/dn, y = dy/dn, z = dz/dn;
  float xx=x*x, yy=y*y, zz=z*z, xy=x*y, yz=y*z, xz=x*z;
  float bas[16];
  bas[0]  =  SH_C0;
  bas[1]  = -SH_C1 * y;
  bas[2]  =  SH_C1 * z;
  bas[3]  = -SH_C1 * x;
  bas[4]  =  1.0925484305920792f  * xy;
  bas[5]  = -1.0925484305920792f  * yz;
  bas[6]  =  0.31539156525252005f * (2.f*zz - xx - yy);
  bas[7]  = -1.0925484305920792f  * xz;
  bas[8]  =  0.5462742152960396f  * (xx - yy);
  bas[9]  = -0.5900435899266435f  * y * (3.f*xx - yy);
  bas[10] =  2.890611442640554f   * xy * z;
  bas[11] = -0.4570457994644658f  * y * (4.f*zz - xx - yy);
  bas[12] =  0.3731763325901154f  * z * (2.f*zz - 3.f*xx - 3.f*yy);
  bas[13] = -0.4570457994644658f  * x * (4.f*zz - xx - yy);
  bas[14] =  1.445305721320277f   * z * (xx - yy);
  bas[15] = -0.5900435899266435f  * x * (xx - 3.f*yy);

  float colr[3];
  #pragma unroll
  for (int ch = 0; ch < 3; ch++) {
    float r = bas[0] * fdc[3*g + ch];
    #pragma unroll
    for (int i = 1; i < 16; i++) r += bas[i] * frest[45*g + (i-1)*3 + ch];
    colr[ch] = fmaxf(r + 0.5f, 0.0f);
  }

  tz_key[g] = tz;
  geo[2*g+0] = make_float4(mx, my, conA, conB);
  geo[2*g+1] = make_float4(conC, valid ? op : 0.0f, tz, 0.0f);
  col[g]     = make_float4(colr[0], colr[1], colr[2], tz);
}

// ---------------------------------------------------------------------------
// Kernel 2: single-block bitonic sort by tz + gather sorted/padded arrays
// ---------------------------------------------------------------------------
#define MAXSORT 4096
__global__ void gs_sort(const float* __restrict__ tz_key,
                        const float4* __restrict__ geo,
                        const float4* __restrict__ col,
                        float4* __restrict__ geo_s,
                        float4* __restrict__ col_s,
                        int N, int npot, int nch)
{
  __shared__ float sk[MAXSORT];
  __shared__ int   sv[MAXSORT];
  int tid = threadIdx.x, B = blockDim.x;
  for (int i = tid; i < npot; i += B) {
    sk[i] = (i < N) ? tz_key[i] : 3.0e38f;
    sv[i] = i;
  }
  __syncthreads();
  for (int k = 2; k <= npot; k <<= 1) {
    for (int j = k >> 1; j > 0; j >>= 1) {
      for (int i = tid; i < npot; i += B) {
        int ixj = i ^ j;
        if (ixj > i) {
          bool up = ((i & k) == 0);
          float ki = sk[i], kx = sk[ixj];
          bool swap = up ? (ki > kx) : (ki < kx);
          if (swap) {
            sk[i] = kx; sk[ixj] = ki;
            int t = sv[i]; sv[i] = sv[ixj]; sv[ixj] = t;
          }
        }
      }
      __syncthreads();
    }
  }
  // gather into sorted arrays, pad to multiple of 32 with alpha=0 entries
  float4 z4 = make_float4(0.f, 0.f, 0.f, 0.f);
  for (int i = tid; i < nch; i += B) {
    if (i < N) {
      int s = sv[i];
      geo_s[2*i+0] = geo[2*s+0];
      geo_s[2*i+1] = geo[2*s+1];
      col_s[i]     = col[s];
    } else {
      geo_s[2*i+0] = z4;
      geo_s[2*i+1] = z4;
      col_s[i]     = z4;
    }
  }
}

// ---------------------------------------------------------------------------
// Kernel 3: WMMA blend. 16 pixels / wave, 32 gaussians / chunk.
//   A (16x32 f16)  = blend weights  w = alpha * T_exclusive
//   B (32x16 f16)  = cols 0..3 = [r,g,b,tz], rest 0
//   C/D (16x16 f32)= running [color|depth] accumulator
//   Chunk staging: async global->LDS double buffer (ASYNCcnt) when available.
// ---------------------------------------------------------------------------
__global__ void __launch_bounds__(256)
gs_blend(const float4* __restrict__ geo_s,
         const float4* __restrict__ col_s,
         const float* __restrict__ bg,
         const int* __restrict__ Wp,
         float* __restrict__ color_out,
         float* __restrict__ depth_out,
         float* __restrict__ amap_out,
         int HW, int nch)
{
  __shared__ float4 geoS[2][64];   // 32 gaussians x 8 floats, double buffered
  __shared__ float4 colS[2][32];   // 32 gaussians x 4 floats

  int tid  = threadIdx.x;
  int lane = tid & 31;
  int wave = tid >> 5;
  bool hi  = lane >= 16;
  int m    = lane & 15;                                    // pixel row in tile
  int tile = (blockIdx.x * 8 + wave) * 16;                 // first pixel
  int pixel = tile + m;
  int Wimg = *Wp;
  float px = (float)(pixel % Wimg);
  float py = (float)(pixel / Wimg);

  int klo = hi ? 8 : 0;      // A-layout: lane m -> K 0-7/16-23, lane m+16 -> K 8-15/24-31
  int nCol = m;              // B/D column = lane&15
  int kbB = hi ? 8 : 0;      // B-layout: lanes 0-15 rows K0-7/16-23, lanes 16-31 rows K8-15/24-31

  float T = 1.0f;            // transmittance, replicated in lanes m and m+16
  v8f acc = {};
  int nchunks = nch >> 5;

#if HAVE_ASYNC_LDS
  // prologue: async-stage chunk 0 into buffer 0
  {
    if (tid < 64) {
      __builtin_amdgcn_global_load_async_to_lds_b128(
          (gptr128_t)(geo_s + tid),
          (lptr128_t)(&geoS[0][tid]), 0, 0);
    } else if (tid < 96) {
      __builtin_amdgcn_global_load_async_to_lds_b128(
          (gptr128_t)(col_s + (tid - 64)),
          (lptr128_t)(&colS[0][tid - 64]), 0, 0);
    }
    __builtin_amdgcn_s_wait_asynccnt(0);
    __syncthreads();
  }
#endif

  for (int cIt = 0; cIt < nchunks; ++cIt) {
#if HAVE_ASYNC_LDS
    int buf = cIt & 1;
    // stage next chunk into the other buffer while this one is consumed;
    // last readers of buf^1 passed the barrier at the end of iteration cIt-1.
    if (cIt + 1 < nchunks) {
      int nb = buf ^ 1;
      const float4* gnext = geo_s + (size_t)(cIt + 1) * 64;
      const float4* cnext = col_s + (size_t)(cIt + 1) * 32;
      if (tid < 64) {
        __builtin_amdgcn_global_load_async_to_lds_b128(
            (gptr128_t)(gnext + tid),
            (lptr128_t)(&geoS[nb][tid]), 0, 0);
      } else if (tid < 96) {
        __builtin_amdgcn_global_load_async_to_lds_b128(
            (gptr128_t)(cnext + (tid - 64)),
            (lptr128_t)(&colS[nb][tid - 64]), 0, 0);
      }
    }
#else
    int buf = 0;
    __syncthreads();                       // previous chunk fully consumed
    if (tid < 64)       geoS[0][tid]      = geo_s[(size_t)cIt * 64 + tid];
    else if (tid < 96)  colS[0][tid - 64] = col_s[(size_t)cIt * 32 + (tid - 64)];
    if (lane == 0 && cIt + 1 < nchunks) {  // gfx1250 global_prefetch_b8
      __builtin_prefetch(geo_s + (size_t)(cIt + 1) * 64, 0, 0);
      __builtin_prefetch(col_s + (size_t)(cIt + 1) * 32, 0, 0);
    }
    __syncthreads();
#endif

    const float* gb = (const float*)&geoS[buf][0];
    const float* cb = (const float*)&colS[buf][0];

    // per-lane alphas for its 16 (pixel,K) slots
    float a[16];
    #pragma unroll
    for (int e = 0; e < 16; e++) {
      int kk = (e < 8) ? (klo + e) : (klo + 16 + (e - 8));
      const float* gp = gb + kk * 8;
      float dxp = px - gp[0], dyp = py - gp[1];
      float power = -0.5f*(gp[2]*dxp*dxp + gp[4]*dyp*dyp) - gp[3]*dxp*dyp;
      float al = fminf(0.99f, gp[5] * __expf(fminf(power, 0.0f)));
      if (power > 0.0f || al < (1.0f/255.0f)) al = 0.0f;
      a[e] = al;
    }

    // exclusive-prefix transmittance: 8-long serial runs + lane-pair exchange
    float plo = 1.0f, phi = 1.0f;
    #pragma unroll
    for (int e = 0; e < 8;  e++) plo *= (1.0f - a[e]);
    #pragma unroll
    for (int e = 8; e < 16; e++) phi *= (1.0f - a[e]);
    float plo_o = __shfl_xor(plo, 16, 32);   // partner group product
    float phi_o = __shfl_xor(phi, 16, 32);
    // K-order group starts: G0(lane lo),G1(lane hi),G2(lane lo),G3(lane hi)
    float start_lo = hi ? (T * plo_o)               : T;
    float start_hi = hi ? (T * plo_o * plo * phi_o) : (T * plo * plo_o);
    T *= plo * plo_o * phi * phi_o;          // identical in both lanes

    v16h Aw;
    float tr = start_lo;
    #pragma unroll
    for (int e = 0; e < 8; e++)  { Aw[e] = (_Float16)(a[e] * tr); tr *= (1.0f - a[e]); }
    tr = start_hi;
    #pragma unroll
    for (int e = 8; e < 16; e++) { Aw[e] = (_Float16)(a[e] * tr); tr *= (1.0f - a[e]); }

    v16h Bv;
    #pragma unroll
    for (int e = 0; e < 16; e++) {
      float v = (nCol < 4) ? cb[((e < 8) ? (kbB + e) : (kbB + 16 + (e - 8)))*4 + nCol] : 0.0f;
      Bv[e] = (_Float16)v;
    }

    // D = A x B + C   (EXEC all-1s here: divergent staging reconverged above)
    acc = __builtin_amdgcn_wmma_f32_16x16x32_f16(
        /*neg_a=*/false, Aw, /*neg_b=*/false, Bv,
        /*c_mod=*/(short)0, acc, /*reuse_a=*/false, /*reuse_b=*/false);

#if HAVE_ASYNC_LDS
    __builtin_amdgcn_s_wait_asynccnt(0);   // next buffer landed (own wave)
    __syncthreads();                       // ...and for all staging waves
#endif
  }

  // epilogue: D layout -> lane holds column nCol for rows M = r + 8*hi
  #pragma unroll
  for (int r = 0; r < 8; r++) {
    int M = r + (hi ? 8 : 0);
    float Tm = __shfl(T, M, 32);            // pixel M's final transmittance
    int p = tile + M;
    if (p < HW) {
      if (nCol < 3)       color_out[nCol*HW + p] = acc[r] + Tm * bg[nCol];
      else if (nCol == 3) depth_out[p] = acc[r];
    }
  }
  if (!hi && pixel < HW) amap_out[pixel] = 1.0f - T;
}

// ---------------------------------------------------------------------------
extern "C" void kernel_launch(void* const* d_in, const int* in_sizes, int n_in,
                              void* d_out, int out_size, void* d_ws, size_t ws_size,
                              hipStream_t stream) {
  const float* xyz      = (const float*)d_in[0];
  const float* scaling  = (const float*)d_in[1];
  const float* rotation = (const float*)d_in[2];
  const float* opacity  = (const float*)d_in[3];
  const float* fdc      = (const float*)d_in[4];
  const float* frest    = (const float*)d_in[5];
  const float* view     = (const float*)d_in[6];
  const float* proj     = (const float*)d_in[7];
  const float* campos   = (const float*)d_in[8];
  const float* bg       = (const float*)d_in[9];
  const int*   Hp       = (const int*)d_in[10];
  const int*   Wp       = (const int*)d_in[11];

  int N  = in_sizes[0] / 3;
  int HW = (out_size - N) / 5;            // out = 3HW color + HW depth + HW amap + N radii
  int nch = ((N + 31) / 32) * 32;         // gaussians padded to chunk multiple
  int npot = 32;
  while (npot < N && npot < MAXSORT) npot <<= 1;

  float* out = (float*)d_out;
  float* color_out = out;
  float* depth_out = out + 3 * (size_t)HW;
  float* amap_out  = out + 4 * (size_t)HW;
  float* radii_out = out + 5 * (size_t)HW;

  // workspace layout (floats; sections kept 16B aligned: N rounded up to 4)
  size_t Na = ((size_t)N + 3) & ~(size_t)3;
  float* wsf     = (float*)d_ws;
  float*  tz_key = wsf;                      // N   floats
  float4* geo    = (float4*)(wsf + Na);      // 8N  floats (2 float4 / gaussian)
  float4* col    = (float4*)(wsf + Na + 8*Na);           // 4N floats
  float4* geo_s  = (float4*)(wsf + Na + 12*Na);          // 8*nch floats
  float4* col_s  = (float4*)(wsf + Na + 12*Na + 8*(size_t)nch);
  (void)ws_size; (void)n_in;

  gs_preprocess<<<(N + 255) / 256, 256, 0, stream>>>(
      xyz, scaling, rotation, opacity, fdc, frest, view, proj, campos,
      Hp, Wp, tz_key, geo, col, radii_out, N);

  gs_sort<<<1, 1024, 0, stream>>>(tz_key, geo, col, geo_s, col_s, N, npot, nch);

  int waves  = (HW + 15) / 16;
  int blocks = (waves + 7) / 8;           // 8 waves (256 threads) per block
  gs_blend<<<blocks, 256, 0, stream>>>(
      geo_s, col_s, bg, Wp, color_out, depth_out, amap_out, HW, nch);
}
